// Encoder_37443524887085
// MI455X (gfx1250) — compile-verified, matrix operand order checked by reading, not run on hardware
//
#include <hip/hip_runtime.h>
#include <cstdint>
#include <cstddef>

// ---------------------------------------------------------------------------
// Types for CDNA5 WMMA (wave32): v_wmma_f32_16x16x32_bf16
// ---------------------------------------------------------------------------
typedef __attribute__((ext_vector_type(16))) __bf16 v16bf;
typedef __attribute__((ext_vector_type(8)))  __bf16 v8bf;
typedef __attribute__((ext_vector_type(8)))  float  v8f;
typedef __attribute__((ext_vector_type(4)))  unsigned int u32x4;
typedef __attribute__((ext_vector_type(8)))  int i32x8;
typedef __attribute__((ext_vector_type(4)))  int i32x4;

namespace {
constexpr int Lc  = 4;
constexpr int Dc  = 1024;
constexpr int Hc  = 16;
constexpr int DKc = 64;
constexpr int FFc = 4096;
constexpr int Sc  = 1024;
constexpr int Bc  = 2;
constexpr int Mc  = Sc * Bc;     // 2048 token rows, row r = s*B + b
constexpr int HD  = Hc * DKc;    // 1024
}

// Assemble a 16-element bf16 fragment from two 16-byte chunks.
// CDNA5 16-bit A/B fragment per lane: lanes 0-15 hold K in {c..c+7, c+16..c+23},
// lanes 16-31 hold K in {c+8..c+15, c+24..c+31} -> caller passes the two chunk
// base pointers (already offset by half*8).
__device__ __forceinline__ v16bf frag16(const __bf16* p0, const __bf16* p1) {
    v8bf lo = *(const v8bf*)p0;
    v8bf hi = *(const v8bf*)p1;
    return __builtin_shufflevector(lo, hi, 0,1,2,3,4,5,6,7,8,9,10,11,12,13,14,15);
}

__device__ __forceinline__ v8f wmma_bf16(v16bf a, v16bf b, v8f c) {
    return __builtin_amdgcn_wmma_f32_16x16x32_bf16(
        /*neg_a=*/false, a, /*neg_b=*/false, b,
        /*c_mod=*/(short)0, c, /*reuse_a=*/false, /*reuse_b=*/false);
}

// ---------------------------------------------------------------------------
// Tensor Data Mover: 2D bf16 tile (tile_d0 elems/row x tile_d1 rows) from
// global (row stride stride0 elems) into LDS at lds_off, with LDS row padding
// (pad_interval/pad_amount are the D# encodings: interval = 2^(code+1) DWORDs,
//  amount = code+1 DWORDs). Descriptor bit packing per CDNA5 ISA.
// ---------------------------------------------------------------------------
__device__ __forceinline__ void tdm_load_2d_bf16(unsigned int lds_off,
                                                 const void* gptr,
                                                 unsigned int tile_d0,
                                                 unsigned int tile_d1,
                                                 unsigned long long stride0,
                                                 unsigned int pad_i_code,
                                                 unsigned int pad_a_code) {
    const unsigned long long ga = (unsigned long long)(uintptr_t)gptr;
    u32x4 g0;
    g0.x = 1u;                                      // count=1, user mode
    g0.y = lds_off;                                 // lds_addr
    g0.z = (unsigned int)ga;                        // global_addr[31:0]
    g0.w = (unsigned int)(ga >> 32) | (2u << 30);   // global_addr[56:32], type=2
    i32x8 g1;
    g1[0] = (int)((1u << 16) |                      // data_size = 2 bytes
                  (1u << 20) |                      // pad_enable
                  (pad_i_code << 22) | (pad_a_code << 25));
    g1[1] = (int)((tile_d0 & 0xFFFFu) << 16);                       // tensor_dim0[15:0]
    g1[2] = (int)(((tile_d0 >> 16) & 0xFFFFu) |                     // tensor_dim0[31:16]
                  ((tile_d1 & 0xFFFFu) << 16));                     // tensor_dim1[15:0]
    g1[3] = (int)(((tile_d1 >> 16) & 0xFFFFu) |                     // tensor_dim1[31:16]
                  ((tile_d0 & 0xFFFFu) << 16));                     // tile_dim0
    g1[4] = (int)(tile_d1 & 0xFFFFu);                               // tile_dim1 (tile_dim2=0)
    g1[5] = (int)(unsigned int)(stride0 & 0xFFFFFFFFull);           // dim0_stride[31:0]
    g1[6] = (int)(unsigned int)((stride0 >> 32) & 0xFFFFull);       // dim0_stride[47:32]
    g1[7] = 0;
    const i32x4 z4 = {0, 0, 0, 0};
#if __clang_major__ >= 23
    const i32x8 z8 = {0, 0, 0, 0, 0, 0, 0, 0};
    __builtin_amdgcn_tensor_load_to_lds(g0, g1, z4, z4, z8, 0);
#else
    __builtin_amdgcn_tensor_load_to_lds(g0, g1, z4, z4, 0);
#endif
}

// ---------------------------------------------------------------------------
// Embedding gather + positional encoding -> x (f32) and xb (bf16)
// ---------------------------------------------------------------------------
__global__ __launch_bounds__(256) void embed_pe(const int* __restrict__ tok,
                                                const float* __restrict__ emb,
                                                float* __restrict__ x,
                                                __bf16* __restrict__ xb) {
    const int r = blockIdx.x;          // token row = s*B + b
    const int s = r / Bc;
    const int t = tok[r];
    const size_t base = (size_t)r * Dc;
    for (int d = threadIdx.x; d < Dc; d += 256) {
        float e    = emb[(size_t)t * Dc + d];
        float expo = 2.0f * (float)d * (1.0f / (float)Dc);
        float feat = __expf(-9.210340371976184f * expo);   // 10000^-expo
        float ang  = (float)s * feat;
        float pe   = ((d & 1) == 0) ? __sinf(ang) : __cosf(ang);
        float v    = e + pe;
        x[base + d]  = v;
        xb[base + d] = (__bf16)v;
    }
}

// ---------------------------------------------------------------------------
// Weight convert + transpose: W[K][N] f32 -> Wt[N][K] bf16 (32x32 LDS tiles)
// ---------------------------------------------------------------------------
__global__ __launch_bounds__(256) void cvt_t(const float* __restrict__ W,
                                             __bf16* __restrict__ Wt,
                                             int K, int N) {
    __shared__ __bf16 tile[32][33];
    const int n0 = blockIdx.x * 32, k0 = blockIdx.y * 32;
    const int tx = threadIdx.x & 31, ty = threadIdx.x >> 5;
    for (int i = ty; i < 32; i += 8)
        tile[i][tx] = (__bf16)W[(size_t)(k0 + i) * N + n0 + tx];
    __syncthreads();
    for (int i = ty; i < 32; i += 8)
        Wt[(size_t)(n0 + i) * K + k0 + tx] = tile[tx][i];
}

// ---------------------------------------------------------------------------
// WMMA GEMM: C[M,N] = A[M,K](bf16, row-major) x Bt[N,K](bf16, K-contiguous)
// Block 256 thr = 8 waves; tile 128x128x32; wave computes 32x64 (2x4 wmma).
// Tiles stream into double-buffered LDS via the Tensor Data Mover: wave 0
// issues tile ki+1 (2 TDM descriptors) then waits TENSORcnt<=2 while all
// waves compute tile ki from the other buffer.
// ---------------------------------------------------------------------------
__global__ __launch_bounds__(256, 1)
void gemm_bf16(const __bf16* __restrict__ A, const __bf16* __restrict__ Bt,
               const float* __restrict__ bias,
               float* __restrict__ Cf, __bf16* __restrict__ Cb,
               int Mdim, int Ndim, int Kdim, int relu) {
    __shared__ __bf16 As[2][128][40];   // 32-wide K slice + 16B pad (TDM pads)
    __shared__ __bf16 Bs[2][128][40];

    const int tid   = threadIdx.x;
    const int wave  = tid >> 5, lane = tid & 31;
    const int l16   = lane & 15, hf = lane >> 4;
    const int m0    = blockIdx.y * 128, n0 = blockIdx.x * 128;
    const int wm    = wave & 3;        // 4 waves along M (32 rows each)
    const int wn    = wave >> 2;       // 2 waves along N (64 cols each)

    const unsigned int aOff[2] = { (unsigned int)(uintptr_t)&As[0][0][0],
                                   (unsigned int)(uintptr_t)&As[1][0][0] };
    const unsigned int bOff[2] = { (unsigned int)(uintptr_t)&Bs[0][0][0],
                                   (unsigned int)(uintptr_t)&Bs[1][0][0] };
    const __bf16* aBase = A  + (size_t)m0 * Kdim;
    const __bf16* bBase = Bt + (size_t)n0 * Kdim;

    v8f acc[2][4];
#pragma unroll
    for (int i = 0; i < 2; ++i)
#pragma unroll
        for (int j = 0; j < 4; ++j) acc[i][j] = {};

    const int nk = Kdim >> 5;
    if (wave == 0) {   // prologue: tile 0 -> buffer 0
        tdm_load_2d_bf16(aOff[0], aBase, 32, 128, (unsigned long long)Kdim, 3, 3);
        tdm_load_2d_bf16(bOff[0], bBase, 32, 128, (unsigned long long)Kdim, 3, 3);
    }

    for (int ki = 0; ki < nk; ++ki) {
        const int buf = ki & 1;
        if (wave == 0) {
            if (ki + 1 < nk) {   // stream next tile into the other buffer
                tdm_load_2d_bf16(aOff[buf ^ 1], aBase + (ki + 1) * 32, 32, 128,
                                 (unsigned long long)Kdim, 3, 3);
                tdm_load_2d_bf16(bOff[buf ^ 1], bBase + (ki + 1) * 32, 32, 128,
                                 (unsigned long long)Kdim, 3, 3);
                __builtin_amdgcn_s_wait_tensorcnt((short)2);  // tile ki complete
            } else {
                __builtin_amdgcn_s_wait_tensorcnt((short)0);
            }
        }
        __syncthreads();

        v16bf af[2], bfrag[4];
#pragma unroll
        for (int tm = 0; tm < 2; ++tm) {
            const int row = wm * 32 + tm * 16 + l16;
            af[tm] = frag16(&As[buf][row][hf * 8], &As[buf][row][16 + hf * 8]);
        }
#pragma unroll
        for (int tn = 0; tn < 4; ++tn) {
            const int col = wn * 64 + tn * 16 + l16;
            bfrag[tn] = frag16(&Bs[buf][col][hf * 8], &Bs[buf][col][16 + hf * 8]);
        }
#pragma unroll
        for (int tm = 0; tm < 2; ++tm)
#pragma unroll
            for (int tn = 0; tn < 4; ++tn)
                acc[tm][tn] = wmma_bf16(af[tm], bfrag[tn], acc[tm][tn]);
        __syncthreads();   // all waves done with buf before it is refilled
    }

    // Epilogue. C layout: VGPR j -> row j (lanes 0-15) / j+8 (lanes 16-31).
#pragma unroll
    for (int tm = 0; tm < 2; ++tm)
#pragma unroll
        for (int tn = 0; tn < 4; ++tn)
#pragma unroll
            for (int j = 0; j < 8; ++j) {
                const int row = m0 + wm * 32 + tm * 16 + j + 8 * hf;
                const int col = n0 + wn * 64 + tn * 16 + l16;
                float v = acc[tm][tn][j];
                if (bias) v += bias[col];
                if (relu) v = fmaxf(v, 0.0f);
                const size_t idx = (size_t)row * Ndim + col;
                if (Cf) Cf[idx] = v;
                if (Cb) Cb[idx] = (__bf16)v;
            }
}

// ---------------------------------------------------------------------------
// Fused flash-style attention. One block = (b, h, 64-query tile); 4 waves,
// each wave owns 16 query rows. Streams 32-key tiles: K tile arrives via the
// Tensor Data Mover, V tile is transposed manually (TDM has no transpose).
// QK^T via 2 chained WMMAs (DK=64), mask + online softmax in registers,
// P->bf16 through wave-private LDS, P.V via 4 WMMAs. Output bf16.
// ---------------------------------------------------------------------------
__global__ __launch_bounds__(128, 1)
void attention(const __bf16* __restrict__ qb, const __bf16* __restrict__ kb,
               const __bf16* __restrict__ vb, const unsigned char* __restrict__ mask,
               __bf16* __restrict__ ob) {
    __shared__ __bf16 kt[32][72];      // [key][dk]  (64 + pad, TDM pads)
    __shared__ __bf16 vt[64][40];      // [dv][key]  (32 + pad) transposed
    __shared__ __bf16 pt[4][16][40];   // per-wave P scratch [qrow][key32]

    const int id   = blockIdx.x;
    const int qblk = (id & 15) * 64;
    const int h    = (id >> 4) & 15;
    const int b    = id >> 8;
    const int tid  = threadIdx.x;
    const int wave = tid >> 5, lane = tid & 31;
    const int l16  = lane & 15, hf = lane >> 4;
    const unsigned int ktOff = (unsigned int)(uintptr_t)&kt[0][0];

    // Q fragments for this wave's 16 rows (DK=64 -> two 16x32 A-fragments)
    const int qs = qblk + wave * 16 + l16;
    const size_t qrow = (size_t)(qs * Bc + b) * HD + h * DKc;
    v16bf aQ[2];
#pragma unroll
    for (int f = 0; f < 2; ++f)
        aQ[f] = frag16(&qb[qrow + f * 32 + hf * 8],
                       &qb[qrow + f * 32 + 16 + hf * 8]);

    v8f acc[4];
#pragma unroll
    for (int n = 0; n < 4; ++n) acc[n] = {};
    float mrun[8], lrun[8];
#pragma unroll
    for (int j = 0; j < 8; ++j) { mrun[j] = -1e30f; lrun[j] = 0.0f; }

    for (int t = 0; t < Sc; t += 32) {
        __syncthreads();
        // K tile via TDM: 32 key rows x 64 dk, row stride B*HD elements,
        // LDS pad 4 DWORDs every 32 DWORDs -> kt[32][72]
        if (wave == 0)
            tdm_load_2d_bf16(ktOff,
                             kb + (size_t)(t * Bc + b) * HD + h * DKc,
                             64, 32, (unsigned long long)(Bc * HD), 4, 3);
        {   // V tile transposed to [dv][key] so P.V B-fragments read
            // contiguous keys per dv column
            const int key = tid >> 2, c = (tid & 3) * 16;
            const size_t src = (size_t)((t + key) * Bc + b) * HD + h * DKc + c;
            v8bf v0 = *(const v8bf*)&vb[src];
            v8bf v1 = *(const v8bf*)&vb[src + 8];
#pragma unroll
            for (int i = 0; i < 8; ++i) {
                vt[c + i][key]     = v0[i];
                vt[c + 8 + i][key] = v1[i];
            }
        }
        if (wave == 0) __builtin_amdgcn_s_wait_tensorcnt((short)0);
        __syncthreads();

        // scores: two 16-key sub-tiles
        v8f c01[2];
#pragma unroll
        for (int sub = 0; sub < 2; ++sub) {
            const int kr = sub * 16 + l16;
            v16bf bk0 = frag16(&kt[kr][hf * 8],      &kt[kr][16 + hf * 8]);
            v16bf bk1 = frag16(&kt[kr][32 + hf * 8], &kt[kr][48 + hf * 8]);
            v8f c = {};
            c = wmma_bf16(aQ[0], bk0, c);
            c = wmma_bf16(aQ[1], bk1, c);
            const int key = t + sub * 16 + l16;
            const bool msk = mask[b * Sc + key] != 0;   // True -> -inf
#pragma unroll
            for (int j = 0; j < 8; ++j)
                c[j] = msk ? -1e30f : c[j] * 0.125f;    // 1/sqrt(64)
            c01[sub] = c;
        }

        // online softmax per row (row = j + 8*hf; reduce across 16-lane half)
#pragma unroll
        for (int j = 0; j < 8; ++j) {
            float mx = fmaxf(c01[0][j], c01[1][j]);
#pragma unroll
            for (int o = 1; o < 16; o <<= 1) mx = fmaxf(mx, __shfl_xor(mx, o, 32));
            const float mnew  = fmaxf(mrun[j], mx);
            const float alpha = __expf(mrun[j] - mnew);
            const float p0 = __expf(c01[0][j] - mnew);
            const float p1 = __expf(c01[1][j] - mnew);
            float ssum = p0 + p1;
#pragma unroll
            for (int o = 1; o < 16; o <<= 1) ssum += __shfl_xor(ssum, o, 32);
            lrun[j] = lrun[j] * alpha + ssum;
            mrun[j] = mnew;
#pragma unroll
            for (int n = 0; n < 4; ++n) acc[n][j] *= alpha;
            c01[0][j] = p0;
            c01[1][j] = p1;
        }

        // P (C-layout f32) -> bf16 A-fragment via wave-private LDS scratch
#pragma unroll
        for (int j = 0; j < 8; ++j) {
            const int prow = j + 8 * hf;
            pt[wave][prow][l16]      = (__bf16)c01[0][j];
            pt[wave][prow][16 + l16] = (__bf16)c01[1][j];
        }
        v16bf aP = frag16(&pt[wave][l16][hf * 8], &pt[wave][l16][16 + hf * 8]);

#pragma unroll
        for (int nt = 0; nt < 4; ++nt) {
            const int dv = nt * 16 + l16;
            v16bf bV = frag16(&vt[dv][hf * 8], &vt[dv][16 + hf * 8]);
            acc[nt] = wmma_bf16(aP, bV, acc[nt]);
        }
    }

    // normalize and store (bf16, feeds the Wo projection GEMM)
#pragma unroll
    for (int j = 0; j < 8; ++j) {
        const float inv = (lrun[j] > 0.0f) ? (1.0f / lrun[j]) : 0.0f;
        const int qsr = qblk + wave * 16 + j + 8 * hf;
        const size_t dst = (size_t)(qsr * Bc + b) * HD + h * DKc;
#pragma unroll
        for (int nt = 0; nt < 4; ++nt)
            ob[dst + nt * 16 + l16] = (__bf16)(acc[nt][j] * inv);
    }
}

// ---------------------------------------------------------------------------
// Residual add + LayerNorm: y = LN(x + d)*g + beta -> xout (f32), xb (bf16)
// ---------------------------------------------------------------------------
__global__ __launch_bounds__(256, 1)
void add_ln(const float* __restrict__ xin, const float* __restrict__ dlt,
            const float* __restrict__ g, const float* __restrict__ beta,
            float* __restrict__ xout, __bf16* __restrict__ xbout) {
    __shared__ float red[8];
    const int r = blockIdx.x, tid = threadIdx.x;
    const size_t base = (size_t)r * Dc;

    float v[4];
    float s = 0.0f;
#pragma unroll
    for (int i = 0; i < 4; ++i) {
        const int d = tid + i * 256;
        v[i] = xin[base + d] + dlt[base + d];
        s += v[i];
    }
#pragma unroll
    for (int o = 1; o < 32; o <<= 1) s += __shfl_xor(s, o, 32);
    if ((tid & 31) == 0) red[tid >> 5] = s;
    __syncthreads();
    float tot = 0.0f;
#pragma unroll
    for (int i = 0; i < 8; ++i) tot += red[i];
    const float mean = tot * (1.0f / Dc);

    float vs = 0.0f;
#pragma unroll
    for (int i = 0; i < 4; ++i) { const float d2 = v[i] - mean; vs += d2 * d2; }
    __syncthreads();
#pragma unroll
    for (int o = 1; o < 32; o <<= 1) vs += __shfl_xor(vs, o, 32);
    if ((tid & 31) == 0) red[tid >> 5] = vs;
    __syncthreads();
    float vtot = 0.0f;
#pragma unroll
    for (int i = 0; i < 8; ++i) vtot += red[i];
    const float rstd = rsqrtf(vtot * (1.0f / Dc) + 1e-5f);

#pragma unroll
    for (int i = 0; i < 4; ++i) {
        const int d = tid + i * 256;
        const float y = (v[i] - mean) * rstd * g[d] + beta[d];
        xout[base + d]  = y;
        xbout[base + d] = (__bf16)y;
    }
}

// ---------------------------------------------------------------------------
// Host launcher
// ---------------------------------------------------------------------------
extern "C" void kernel_launch(void* const* d_in, const int* in_sizes, int n_in,
                              void* d_out, int out_size, void* d_ws, size_t ws_size,
                              hipStream_t stream) {
    (void)in_sizes; (void)n_in; (void)out_size; (void)ws_size;

    const int*           tokens = (const int*)d_in[0];
    const unsigned char* maskp  = (const unsigned char*)d_in[1];   // bool [B,1,1,S]
    const float* emb   = (const float*)d_in[2];
    const float* Wq    = (const float*)d_in[3];
    const float* Wk    = (const float*)d_in[4];
    const float* Wv    = (const float*)d_in[5];
    const float* Wo    = (const float*)d_in[6];
    const float* ln1g  = (const float*)d_in[7];
    const float* ln1b  = (const float*)d_in[8];
    const float* W1    = (const float*)d_in[9];
    const float* b1    = (const float*)d_in[10];
    const float* W2    = (const float*)d_in[11];
    const float* b2    = (const float*)d_in[12];
    const float* ln2g  = (const float*)d_in[13];
    const float* ln2b  = (const float*)d_in[14];

    char* p = (char*)d_ws;
    auto take = [&](size_t bytes) -> char* {
        char* r = p;
        p += (bytes + 255) & ~(size_t)255;
        return r;
    };
    float*  x   = (float*)take((size_t)Mc * Dc * 4);   // residual stream f32
    __bf16* xb  = (__bf16*)take((size_t)Mc * Dc * 2);  // residual stream bf16
    __bf16* qb  = (__bf16*)take((size_t)Mc * HD * 2);
    __bf16* kb  = (__bf16*)take((size_t)Mc * HD * 2);
    __bf16* vb  = (__bf16*)take((size_t)Mc * HD * 2);
    __bf16* ob  = (__bf16*)take((size_t)Mc * HD * 2);
    float*  t0  = (float*)take((size_t)Mc * Dc * 4);   // GEMM f32 staging
    __bf16* ffb = (__bf16*)take((size_t)Mc * FFc * 2);
    __bf16* Wqt = (__bf16*)take((size_t)Dc * HD * 2);  // per-layer bf16 weights
    __bf16* Wkt = (__bf16*)take((size_t)Dc * HD * 2);
    __bf16* Wvt = (__bf16*)take((size_t)Dc * HD * 2);
    __bf16* Wot = (__bf16*)take((size_t)HD * Dc * 2);
    __bf16* W1t = (__bf16*)take((size_t)Dc * FFc * 2);
    __bf16* W2t = (__bf16*)take((size_t)FFc * Dc * 2);

    embed_pe<<<Mc, 256, 0, stream>>>(tokens, emb, x, xb);

    const dim3 gProj(HD / 128, Mc / 128);    // N=1024
    const dim3 gFF1(FFc / 128, Mc / 128);    // N=4096
    const dim3 gFF2(Dc / 128, Mc / 128);     // N=1024

    for (int l = 0; l < Lc; ++l) {
        const float* wq = Wq + (size_t)l * Dc * HD;
        const float* wk = Wk + (size_t)l * Dc * HD;
        const float* wv = Wv + (size_t)l * Dc * HD;
        const float* wo = Wo + (size_t)l * HD * Dc;
        const float* w1 = W1 + (size_t)l * Dc * FFc;
        const float* w2 = W2 + (size_t)l * FFc * Dc;

        // stage this layer's weights as bf16 [N][K]
        cvt_t<<<dim3(HD / 32, Dc / 32), 256, 0, stream>>>(wq, Wqt, Dc, HD);
        cvt_t<<<dim3(HD / 32, Dc / 32), 256, 0, stream>>>(wk, Wkt, Dc, HD);
        cvt_t<<<dim3(HD / 32, Dc / 32), 256, 0, stream>>>(wv, Wvt, Dc, HD);
        cvt_t<<<dim3(Dc / 32, HD / 32), 256, 0, stream>>>(wo, Wot, HD, Dc);
        cvt_t<<<dim3(FFc / 32, Dc / 32), 256, 0, stream>>>(w1, W1t, Dc, FFc);
        cvt_t<<<dim3(Dc / 32, FFc / 32), 256, 0, stream>>>(w2, W2t, FFc, Dc);

        // QKV projections (bf16 outputs feed attention WMMAs)
        gemm_bf16<<<gProj, 256, 0, stream>>>(xb, Wqt, nullptr, nullptr, qb, Mc, HD, Dc, 0);
        gemm_bf16<<<gProj, 256, 0, stream>>>(xb, Wkt, nullptr, nullptr, kb, Mc, HD, Dc, 0);
        gemm_bf16<<<gProj, 256, 0, stream>>>(xb, Wvt, nullptr, nullptr, vb, Mc, HD, Dc, 0);

        // fused masked-softmax attention
        attention<<<Bc * Hc * (Sc / 64), 128, 0, stream>>>(qb, kb, vb, maskp, ob);

        // output projection + residual LN
        gemm_bf16<<<gFF2, 256, 0, stream>>>(ob, Wot, nullptr, t0, nullptr, Mc, Dc, HD, 0);
        add_ln<<<Mc, 256, 0, stream>>>(x, t0, ln1g + (size_t)l * Dc, ln1b + (size_t)l * Dc, x, xb);

        // FFN
        gemm_bf16<<<gFF1, 256, 0, stream>>>(xb, W1t, b1 + (size_t)l * FFc, nullptr, ffb, Mc, FFc, Dc, 1);
        gemm_bf16<<<gFF2, 256, 0, stream>>>(ffb, W2t, b2 + (size_t)l * Dc, t0, nullptr, Mc, Dc, FFc, 0);

        float* xo = (l == Lc - 1) ? (float*)d_out : x;
        add_ln<<<Mc, 256, 0, stream>>>(x, t0, ln2g + (size_t)l * Dc, ln2b + (size_t)l * Dc, xo, xb);
    }
}
